// AttentionBlock_5506148074045
// MI455X (gfx1250) — matrix operand order repaired and checked by reading, not compiled, not run on hardware
//
#include <hip/hip_runtime.h>
#include <hip/hip_bf16.h>

// ---------------------------------------------------------------------------
// AttentionBlock for MI455X (gfx1250): bf16 WMMA everywhere, f32 accumulate.
// Attention K/V tiles staged into LDS by the Tensor Data Mover (TDM) with
// double buffering, tracked by TENSORcnt.
// ---------------------------------------------------------------------------

typedef __attribute__((ext_vector_type(16))) __bf16 bf16x16;
typedef __attribute__((ext_vector_type(8)))  __bf16 bf16x8;
typedef __attribute__((ext_vector_type(8)))  float  v8f;
typedef __attribute__((ext_vector_type(4)))  unsigned int u32x4;
typedef __attribute__((ext_vector_type(8)))  int i32x8;
typedef __attribute__((ext_vector_type(4)))  int i32x4;

#define DEVINL __device__ __forceinline__

#define HAVE_TDM __has_builtin(__builtin_amdgcn_tensor_load_to_lds)

DEVINL v8f vzero() {
  v8f v;
#pragma unroll
  for (int i = 0; i < 8; ++i) v[i] = 0.0f;
  return v;
}

// A-matrix fragment (16x32 bf16, MxK) from a row-major [m][k] tile.
// ISA layout: lane m = l&15; lanes<16 hold K {0..7,16..23}, lanes>=16 {8..15,24..31}.
DEVINL bf16x16 frag_a(const __bf16* base, int stride, int lane) {
  const int m = lane & 15;
  const int koff = (lane >> 4) * 8;          // elements
  const __bf16* p = base + m * stride + koff;
  bf16x8 lo = *(const bf16x8*)(p);           // K [koff .. koff+7]
  bf16x8 hi = *(const bf16x8*)(p + 16);      // K [koff+16 .. koff+23]
  bf16x16 r;
#pragma unroll
  for (int i = 0; i < 8; ++i) { r[i] = lo[i]; r[i + 8] = hi[i]; }
  return r;
}

// B-matrix fragment (32x16 bf16, KxN) from a row-major [n][k] tile (i.e. B^T).
// ISA layout: lane n = l&15; lanes<16 hold K {0..7,16..23}, lanes>=16 {8..15,24..31}.
DEVINL bf16x16 frag_b(const __bf16* base, int stride, int lane) {
  const int n = lane & 15;
  const int koff = (lane >> 4) * 8;          // elements
  const __bf16* p = base + n * stride + koff;
  bf16x8 lo = *(const bf16x8*)(p);
  bf16x8 hi = *(const bf16x8*)(p + 16);
  bf16x16 r;
#pragma unroll
  for (int i = 0; i < 8; ++i) { r[i] = lo[i]; r[i + 8] = hi[i]; }
  return r;
}

DEVINL v8f wmma_bf16(bf16x16 a, bf16x16 b, v8f c) {
  return __builtin_amdgcn_wmma_f32_16x16x32_bf16(
      false, a, false, b, (short)0, c, false, false);
}

// Low 32 bits of a generic LDS pointer == byte offset in the wave's LDS space.
DEVINL unsigned int lds_offset(const void* p) {
  return (unsigned int)(uintptr_t)p;
}

#if HAVE_TDM
// 2D tile DMA: global (row-major, row_stride elements of bf16) -> LDS, with
// hardware row padding so LDS rows land at (tile_w + pad) element stride.
// Descriptor packing per CDNA5 ISA sect. 8.3-8.6.
// pad_int_code: pad every 8<<code bytes; pad_amt_code: insert 4*(code+1) bytes.
DEVINL void tdm_load_tile_2d(unsigned int lds_off, const void* gsrc,
                             unsigned int tile_w, unsigned int tile_h,
                             unsigned int tensor_w, unsigned int tensor_h,
                             unsigned int row_stride, unsigned int pad_int_code,
                             unsigned int pad_amt_code) {
  const unsigned long long ga = (unsigned long long)(uintptr_t)gsrc;
  u32x4 g0;
  g0[0] = 1u;                                               // count=1, user desc
  g0[1] = lds_off;                                          // lds_addr
  g0[2] = (unsigned int)ga;                                 // global_addr[31:0]
  g0[3] = ((unsigned int)(ga >> 32) & 0x01FFFFFFu) | (2u << 30);  // addr[56:32], type=2
  i32x8 g1;
  g1[0] = (int)((1u << 16)                                  // data_size = 2B (bf16)
                | (1u << 20)                                // pad_enable
                | (pad_int_code << 22) | (pad_amt_code << 25));
  g1[1] = (int)((tensor_w & 0xFFFFu) << 16);                // tensor_dim0[15:0]
  g1[2] = (int)((tensor_w >> 16) | ((tensor_h & 0xFFFFu) << 16));
  g1[3] = (int)((tensor_h >> 16) | ((tile_w & 0xFFFFu) << 16));
  g1[4] = (int)(tile_h & 0xFFFFu);                          // tile_dim1 (tile_dim2=0)
  g1[5] = (int)row_stride;                                  // tensor_dim0_stride[31:0]
  g1[6] = 0;
  g1[7] = 0;
  i32x4 gz;
  gz[0] = gz[1] = gz[2] = gz[3] = 0;
#if defined(__clang_major__) && (__clang_major__ >= 23)
  i32x8 gz8;
#pragma unroll
  for (int i = 0; i < 8; ++i) gz8[i] = 0;
  __builtin_amdgcn_tensor_load_to_lds(g0, g1, gz, gz, gz8, 0);
#else
  __builtin_amdgcn_tensor_load_to_lds(g0, g1, gz, gz, 0);
#endif
}

DEVINL void wait_tensorcnt_le2() {
#if __has_builtin(__builtin_amdgcn_s_wait_tensorcnt)
  __builtin_amdgcn_s_wait_tensorcnt(2);
#else
  asm volatile("s_wait_tensorcnt 0x2" ::: "memory");
#endif
}
DEVINL void wait_tensorcnt_le0() {
#if __has_builtin(__builtin_amdgcn_s_wait_tensorcnt)
  __builtin_amdgcn_s_wait_tensorcnt(0);
#else
  asm volatile("s_wait_tensorcnt 0x0" ::: "memory");
#endif
}
#endif  // HAVE_TDM

// ---------------------------------------------------------------------------
// Kernel 0: convert weight matrices f32 -> bf16 (one-time, tiny).
// ---------------------------------------------------------------------------
__global__ void cvt_weights_kernel(const float* __restrict__ wqkv,
                                   const float* __restrict__ wout,
                                   __bf16* __restrict__ wqkv_bf,
                                   __bf16* __restrict__ wout_bf) {
  int i = blockIdx.x * blockDim.x + threadIdx.x;
  if (i < 768 * 256) wqkv_bf[i] = (__bf16)wqkv[i];
  if (i < 256 * 256) wout_bf[i] = (__bf16)wout[i];
}

// ---------------------------------------------------------------------------
// Kernel 1: QKV projection. Per batch: QKV(768,4096) = W(768,256) @ X(256,4096).
// Output: Q,K as bf16 [b][n][c] (c contiguous), V as bf16 [b][c][n] (transposed).
// ---------------------------------------------------------------------------
__global__ __launch_bounds__(256) void qkv_kernel(
    const float* __restrict__ x, const __bf16* __restrict__ wbf,
    const float* __restrict__ bqkv, __bf16* __restrict__ Qb,
    __bf16* __restrict__ Kb, __bf16* __restrict__ Vt) {
  __shared__ __bf16 sX[64 * 264];                 // x tile transposed [n][c], padded
  const int b = blockIdx.y;
  const int n0 = blockIdx.x * 64;
  const int tid = threadIdx.x;
  const int lane = tid & 31;
  const int wave = tid >> 5;
  const int h = lane >> 4;
  const float* xb = x + ((size_t)b << 20);        // 256*4096 per batch

  for (int idx = tid; idx < 64 * 256; idx += 256) {
    int c = idx >> 6, n = idx & 63;
    sX[n * 264 + c] = (__bf16)xb[(c << 12) + n0 + n];
  }
  __syncthreads();

  for (int mc = 0; mc < 6; ++mc) {                // 768 output rows in 6x128 chunks
    const int m0 = mc * 128 + wave * 16;
    v8f acc[4];
#pragma unroll
    for (int j = 0; j < 4; ++j) acc[j] = vzero();

    for (int k0 = 0; k0 < 256; k0 += 32) {
      bf16x16 a = frag_a(wbf + m0 * 256 + k0, 256, lane);
#pragma unroll
      for (int j = 0; j < 4; ++j) {
        bf16x16 bfr = frag_b(sX + j * 16 * 264 + k0, 264, lane);
        acc[j] = wmma_bf16(a, bfr, acc[j]);
      }
    }

    float bias[8];
#pragma unroll
    for (int r = 0; r < 8; ++r) bias[r] = bqkv[m0 + r + 8 * h];
    const int c0 = m0 & 255;
    const int sel = m0 >> 8;                      // 0=Q 1=K 2=V
#pragma unroll
    for (int j = 0; j < 4; ++j) {
      const int n = n0 + j * 16 + (lane & 15);
      if (sel < 2) {
        __bf16* dst = (sel == 0 ? Qb : Kb) +
                      (((size_t)(b * 4096 + n)) << 8) + c0 + 8 * h;
        bf16x8 v;
#pragma unroll
        for (int r = 0; r < 8; ++r) v[r] = (__bf16)(acc[j][r] + bias[r]);
        *(bf16x8*)dst = v;                        // 16B packed store (8 channels)
      } else {
#pragma unroll
        for (int r = 0; r < 8; ++r) {
          const int c = c0 + r + 8 * h;
          Vt[((size_t)b << 20) + ((size_t)c << 12) + n] =
              (__bf16)(acc[j][r] + bias[r]);
        }
      }
    }
  }
}

// ---------------------------------------------------------------------------
// Kernel 2: flash attention with TDM-staged, double-buffered K/V tiles.
// Grid (32,4) x 256 threads; each wave owns 16 query rows.
// ---------------------------------------------------------------------------
__global__ __launch_bounds__(256) void attn_kernel(
    const __bf16* __restrict__ Qb, const __bf16* __restrict__ Kb,
    const __bf16* __restrict__ Vt, __bf16* __restrict__ Ob) {
  __shared__ __bf16 sK[2][32 * 264];              // key rows [nk][c], padded
  __shared__ __bf16 sV[2][256 * 40];              // V^T [c][nk], padded
  __shared__ __bf16 sP[8 * 16 * 40];              // per-wave P [m][nk]
  const int b = blockIdx.y;
  const int tid = threadIdx.x;
  const int lane = tid & 31;
  const int wave = tid >> 5;
  const int h = lane >> 4;
  const int nq0 = blockIdx.x * 128 + wave * 16;
  const size_t base = (size_t)b * 4096 * 256;
  const __bf16* Kbb = Kb + base;
  const __bf16* Vtb = Vt + base;

  // Q fragments for this wave's 16 rows, all of K=256 (8 chunks of 32).
  bf16x16 qf[8];
#pragma unroll
  for (int kc = 0; kc < 8; ++kc)
    qf[kc] = frag_a(Qb + base + (size_t)nq0 * 256 + kc * 32, 256, lane);

  v8f oacc[16];
#pragma unroll
  for (int ct = 0; ct < 16; ++ct) oacc[ct] = vzero();
  float rowmax[8], rowsum[8];
#pragma unroll
  for (int r = 0; r < 8; ++r) { rowmax[r] = -1e30f; rowsum[r] = 0.0f; }

#if HAVE_TDM
  if (wave == 0) {
    // K tile: 32 rows x 256 elems (512B rows), pad 16B each 512B -> stride 264.
    tdm_load_tile_2d(lds_offset(&sK[0][0]), Kbb, 256, 32, 256, 4096, 256, 6, 3);
    // V^T tile: 256 rows x 32 elems (64B rows), pad 16B each 64B -> stride 40.
    tdm_load_tile_2d(lds_offset(&sV[0][0]), Vtb, 32, 256, 4096, 256, 4096, 3, 3);
  }
#endif

  for (int it = 0; it < 128; ++it) {
    const int kt = it << 5;
    const int cur = it & 1;
#if HAVE_TDM
    __syncthreads();  // everyone done reading the back buffer (WAR vs next DMA)
    if (wave == 0) {
      if (it + 1 < 128) {
        const int ktn = kt + 32;
        tdm_load_tile_2d(lds_offset(&sK[cur ^ 1][0]),
                         Kbb + (size_t)ktn * 256, 256, 32, 256, 4096 - ktn,
                         256, 6, 3);
        tdm_load_tile_2d(lds_offset(&sV[cur ^ 1][0]),
                         Vtb + ktn, 32, 256, 4096 - ktn, 256, 4096, 3, 3);
        wait_tensorcnt_le2();  // current tile's 2 DMAs retired (in-order)
      } else {
        wait_tensorcnt_le0();
      }
    }
    __syncthreads();  // current tile visible to all waves
#else
    __syncthreads();
    for (int idx = tid; idx < 1024; idx += 256) {
      int row = idx >> 5, chunk = idx & 31;
      *(bf16x8*)(&sK[cur][0] + row * 264 + chunk * 8) =
          *(const bf16x8*)(Kbb + (size_t)(kt + row) * 256 + chunk * 8);
    }
    for (int idx = tid; idx < 1024; idx += 256) {
      int c = idx >> 2, chunk = idx & 3;
      *(bf16x8*)(&sV[cur][0] + c * 40 + chunk * 8) =
          *(const bf16x8*)(Vtb + ((size_t)c << 12) + kt + chunk * 8);
    }
    __syncthreads();
#endif

    // S = (Q K^T) * C^-0.5 for two 16x16 key tiles.
    v8f s0 = vzero(), s1 = vzero();
#pragma unroll
    for (int kc = 0; kc < 8; ++kc) {
      s0 = wmma_bf16(qf[kc], frag_b(&sK[cur][0] + kc * 32, 264, lane), s0);
      s1 = wmma_bf16(qf[kc], frag_b(&sK[cur][0] + 16 * 264 + kc * 32, 264, lane), s1);
    }

    // Online softmax. Row m = r + 8h lives in VGPR r of this lane half; the 16
    // key columns are spread over the half's lanes -> xor reductions 1/2/4/8.
    __bf16* pw = sP + wave * 16 * 40;
#pragma unroll
    for (int r = 0; r < 8; ++r) {
      float a0 = s0[r] * 0.0625f, a1 = s1[r] * 0.0625f;
      float vmax = fmaxf(a0, a1);
      vmax = fmaxf(vmax, __shfl_xor(vmax, 1));
      vmax = fmaxf(vmax, __shfl_xor(vmax, 2));
      vmax = fmaxf(vmax, __shfl_xor(vmax, 4));
      vmax = fmaxf(vmax, __shfl_xor(vmax, 8));
      float newmax = fmaxf(rowmax[r], vmax);
      float alpha = __expf(rowmax[r] - newmax);
      rowmax[r] = newmax;
      float p0 = __expf(a0 - newmax);
      float p1 = __expf(a1 - newmax);
      float rs = p0 + p1;
      rs += __shfl_xor(rs, 1);
      rs += __shfl_xor(rs, 2);
      rs += __shfl_xor(rs, 4);
      rs += __shfl_xor(rs, 8);
      rowsum[r] = rowsum[r] * alpha + rs;
#pragma unroll
      for (int ct = 0; ct < 16; ++ct) oacc[ct][r] *= alpha;
      const int m = r + 8 * h;
      pw[m * 40 + (lane & 15)] = (__bf16)p0;
      pw[m * 40 + 16 + (lane & 15)] = (__bf16)p1;
    }
    __syncthreads();                              // P visible for transpose read

    bf16x16 pa = frag_a(pw, 40, lane);            // P as A fragment (16x32)
#pragma unroll
    for (int ct = 0; ct < 16; ++ct)
      oacc[ct] = wmma_bf16(pa, frag_b(&sV[cur][0] + ct * 16 * 40, 40, lane),
                           oacc[ct]);
  }

  // Normalize and write O[b][n][c].
#pragma unroll
  for (int ct = 0; ct < 16; ++ct) {
    const int c = ct * 16 + (lane & 15);
#pragma unroll
    for (int r = 0; r < 8; ++r) {
      const int n = nq0 + r + 8 * h;
      Ob[base + (size_t)n * 256 + c] = (__bf16)(oacc[ct][r] / rowsum[r]);
    }
  }
}

// ---------------------------------------------------------------------------
// Kernel 3: out = Wout @ O + bias + x (residual), f32 output.
// ---------------------------------------------------------------------------
__global__ __launch_bounds__(256) void proj_kernel(
    const __bf16* __restrict__ Ob, const __bf16* __restrict__ wbf,
    const float* __restrict__ bout, const float* __restrict__ x,
    float* __restrict__ out) {
  const int b = blockIdx.y;
  const int n0 = blockIdx.x * 64;
  const int tid = threadIdx.x;
  const int lane = tid & 31;
  const int wave = tid >> 5;
  const int h = lane >> 4;
  const size_t base = (size_t)b * 4096 * 256;

  // Prefetch the residual lines this wave will read in the epilogue.
#pragma unroll
  for (int j = 0; j < 4; ++j)
    __builtin_prefetch(x + (((size_t)(b * 256 + wave * 32)) << 12) + n0 + j * 16 + lane,
                       0, 1);

  v8f acc[2][4];
#pragma unroll
  for (int mi = 0; mi < 2; ++mi)
#pragma unroll
    for (int j = 0; j < 4; ++j) acc[mi][j] = vzero();

  for (int k0 = 0; k0 < 256; k0 += 32) {
    bf16x16 a0 = frag_a(wbf + (wave * 32) * 256 + k0, 256, lane);
    bf16x16 a1 = frag_a(wbf + (wave * 32 + 16) * 256 + k0, 256, lane);
#pragma unroll
    for (int j = 0; j < 4; ++j) {
      bf16x16 bfr = frag_b(Ob + base + (size_t)(n0 + j * 16) * 256 + k0, 256, lane);
      acc[0][j] = wmma_bf16(a0, bfr, acc[0][j]);
      acc[1][j] = wmma_bf16(a1, bfr, acc[1][j]);
    }
  }

#pragma unroll
  for (int mi = 0; mi < 2; ++mi) {
#pragma unroll
    for (int r = 0; r < 8; ++r) {
      const int o = wave * 32 + mi * 16 + r + 8 * h;
      const float bias = bout[o];
#pragma unroll
      for (int j = 0; j < 4; ++j) {
        const int n = n0 + j * 16 + (lane & 15);
        const size_t idx = (((size_t)(b * 256 + o)) << 12) + n;
        out[idx] = acc[mi][j][r] + bias + x[idx];
      }
    }
  }
}

// ---------------------------------------------------------------------------
extern "C" void kernel_launch(void* const* d_in, const int* in_sizes, int n_in,
                              void* d_out, int out_size, void* d_ws,
                              size_t ws_size, hipStream_t stream) {
  const float* x    = (const float*)d_in[0];
  const float* wqkv = (const float*)d_in[1];
  const float* bqkv = (const float*)d_in[2];
  const float* wout = (const float*)d_in[3];
  const float* bout = (const float*)d_in[4];
  float* out = (float*)d_out;

  char* ws = (char*)d_ws;
  __bf16* wqkv_bf = (__bf16*)ws; ws += (size_t)768 * 256 * 2;
  __bf16* wout_bf = (__bf16*)ws; ws += (size_t)256 * 256 * 2;
  const size_t tensor_bytes = (size_t)4 * 4096 * 256 * 2;   // 8.39 MB each
  __bf16* Qb = (__bf16*)ws; ws += tensor_bytes;
  __bf16* Kb = (__bf16*)ws; ws += tensor_bytes;
  __bf16* Vt = (__bf16*)ws; ws += tensor_bytes;
  __bf16* Ob = (__bf16*)ws; ws += tensor_bytes;

  cvt_weights_kernel<<<768, 256, 0, stream>>>(wqkv, wout, wqkv_bf, wout_bf);
  qkv_kernel<<<dim3(64, 4), 256, 0, stream>>>(x, wqkv_bf, bqkv, Qb, Kb, Vt);
  attn_kernel<<<dim3(32, 4), 256, 0, stream>>>(Qb, Kb, Vt, Ob);
  proj_kernel<<<dim3(64, 4), 256, 0, stream>>>(Ob, wout_bf, bout, x, out);
}
